// FixedSparseMoELLM_71545565217044
// MI455X (gfx1250) — compile-verified, matrix operand order checked by reading, not run on hardware
//
#include <hip/hip_runtime.h>
#include <hip/hip_bf16.h>

// ---------------- model dims ----------------
constexpr int kL  = 2;
constexpr int kD  = 1024;
constexpr int kH  = 16;
constexpr int kDH = 64;
constexpr int kFF = 4096;
constexpr int kE  = 8;
constexpr int kV  = 32000;
constexpr int kIH = 4;
constexpr int kID = 64;
constexpr int kB  = 2;
constexpr int kS  = 1024;
constexpr int kN  = kB * kS;          // 2048 tokens
constexpr float kNEG = -1e9f;

// ---------------- WMMA / TDM types ----------------
typedef __attribute__((ext_vector_type(16))) __bf16 v16bf;
typedef __attribute__((ext_vector_type(8)))  float  v8f;
typedef __attribute__((ext_vector_type(4)))  unsigned int u32x4;
typedef __attribute__((ext_vector_type(8)))  int          i32x8;
typedef __attribute__((ext_vector_type(4)))  int          i32x4;

__device__ __forceinline__ v8f wmma_bf16(v16bf a, v16bf b, v8f c) {
  // D = A(16x32 bf16) x B(32x16 bf16) + C(16x16 f32)
  return __builtin_amdgcn_wmma_f32_16x16x32_bf16(
      false, a, false, b, (short)0, c, false, false);
}

// Load one 16x32 bf16 fragment (A or B^T) for this lane from an LDS row whose
// K-dim is contiguous. lanes 0-15 need K {0..7,16..23}; lanes 16-31 need
// K {8..15,24..31} -> two 16-byte ds_load_b128.
__device__ __forceinline__ v16bf ld_frag(const __bf16* row, int hi) {
  union { v16bf v; uint4 q[2]; } u;
  u.q[0] = *reinterpret_cast<const uint4*>(row + hi * 8);
  u.q[1] = *reinterpret_cast<const uint4*>(row + 16 + hi * 8);
  return u.v;
}

__device__ __forceinline__ float gelu_tanh(float x) {
  return 0.5f * x * (1.0f + tanhf(0.7978845608028654f * (x + 0.044715f * x * x * x)));
}

// ---------------- weight transpose+convert: W[K][N] f32 -> Wt[N][K] bf16 ------
__global__ __launch_bounds__(256) void transpose_to_bf16_kernel(
    const float* __restrict__ W, __bf16* __restrict__ Wt, int K, int Nn)
{
  __shared__ float tile[32][33];
  int kb = blockIdx.y * 32, nb = blockIdx.x * 32;
  int tx = threadIdx.x & 31, ty = threadIdx.x >> 5;   // 8 rows of 32 lanes
  for (int r = 0; r < 4; ++r) {
    int k = kb + ty + r * 8, n = nb + tx;
    tile[ty + r * 8][tx] = (k < K && n < Nn) ? W[(size_t)k * Nn + n] : 0.f;
  }
  __syncthreads();
  for (int r = 0; r < 4; ++r) {
    int n = nb + ty + r * 8, k = kb + tx;
    if (n < Nn && k < K) Wt[(size_t)n * K + k] = (__bf16)tile[tx][ty + r * 8];
  }
}

// ---------------- generic bf16-WMMA GEMM with TDM B-tile staging ----------------
// C[M,N] (f32) = epilogue(A[M,K] (f32) x Bt^T + bias), Bt is bf16 [N][K].
enum { EPI_STORE = 0, EPI_ADD = 1, EPI_GELU = 2, EPI_GATED_ADD = 3 };

#define GEMM_BM 128
#define GEMM_BN 128
#define GEMM_BK 32

__global__ __launch_bounds__(256) void gemm_bf16_kernel(
    const float* __restrict__ A, const __bf16* __restrict__ Bt,
    const float* __restrict__ bias, float* __restrict__ C,
    const float* __restrict__ gate, int gstride,
    int M, int Nn, int K, int mode)
{
  __shared__ __align__(16) __bf16 As[GEMM_BM * GEMM_BK];   // [M][K]
  __shared__ __align__(16) __bf16 Bs[GEMM_BN * GEMM_BK];   // [N][K]

  const int tid  = threadIdx.x;
  const int m0   = blockIdx.y * GEMM_BM;
  const int n0   = blockIdx.x * GEMM_BN;
  const int w    = tid >> 5;
  const int lane = tid & 31;
  const int hi   = lane >> 4;
  const int rw   = lane & 15;
  const int wm   = w >> 1;   // 0..3 -> 32-row slab
  const int wn   = w & 1;    // 0..1 -> 64-col slab

  v8f acc[2][4];
  for (int i = 0; i < 2; ++i)
    for (int j = 0; j < 4; ++j)
      acc[i][j] = (v8f){0.f,0.f,0.f,0.f,0.f,0.f,0.f,0.f};

  for (int kb = 0; kb < K; kb += GEMM_BK) {
#if defined(__AMDGCN__) && __has_builtin(__builtin_amdgcn_tensor_load_to_lds)
    // Tensor Data Mover: DMA the 128x32 bf16 B tile (8KB, X-fastest fill order
    // == our row-major [n][k] layout) straight from global into LDS.
    if (w == 0) {
      unsigned long long ga =
          (unsigned long long)(const void*)(Bt + (size_t)n0 * K + kb);
      unsigned ldsoff = (unsigned)(unsigned long long)(const void*)(&Bs[0]);
      int dim0 = K - kb;        // remaining contiguous extent from tile start
      int dim1 = Nn - n0;       // remaining rows from tile start
      u32x4 g0;
      g0[0] = 1u;                                           // count=1, user D#
      g0[1] = ldsoff;                                       // lds_addr
      g0[2] = (unsigned)(ga & 0xFFFFFFFFull);               // global_addr lo
      g0[3] = (unsigned)((ga >> 32) & 0x1FFFFFFull)         // global_addr hi
              | (2u << 30);                                 // type = 2 (image)
      i32x8 g1;
      g1[0] = (int)(1u << 16);                              // data_size=1 (2B)
      g1[1] = (int)(((unsigned)dim0 & 0xFFFFu) << 16);      // tensor_dim0 lo16
      g1[2] = (int)((((unsigned)dim0 >> 16) & 0xFFFFu)      // tensor_dim0 hi16
              | (((unsigned)dim1 & 0xFFFFu) << 16));        // tensor_dim1 lo16
      g1[3] = (int)((((unsigned)dim1 >> 16) & 0xFFFFu)      // tensor_dim1 hi16
              | (32u << 16));                               // tile_dim0 = 32
      g1[4] = 128;                                          // tile_dim1 = 128
      g1[5] = K;                                            // dim0_stride lo32 (elements)
      g1[6] = 0;
      g1[7] = 0;
      i32x4 gz4 = (i32x4){0, 0, 0, 0};                      // 2D tensor: groups 2/3 null
      i32x8 gz8 = (i32x8){0, 0, 0, 0, 0, 0, 0, 0};
      __builtin_amdgcn_tensor_load_to_lds(g0, g1, gz4, gz4, gz8, 0);
    }
#else
    // fallback: manual bf16 copy of the B tile
    for (int r = 0; r < 16; ++r) {
      int lin = tid + r * 256;
      int nn = lin >> 5, kk = lin & 31;
      __bf16 v = (__bf16)0.f;
      if (n0 + nn < Nn && kb + kk < K) v = Bt[(size_t)(n0 + nn) * K + kb + kk];
      Bs[lin] = v;
    }
#endif
    // stage A tile: contiguous f32 reads along K, convert to bf16
    for (int r = 0; r < 4; ++r) {
      int lin = (tid + r * 256) * 4;
      int mm = lin / GEMM_BK, kk = lin % GEMM_BK;
      int gm = m0 + mm, gk = kb + kk;
      float4 v = {0.f, 0.f, 0.f, 0.f};
      if (gm < M && gk < K)
        v = *reinterpret_cast<const float4*>(A + (size_t)gm * K + gk);
      As[mm * GEMM_BK + kk + 0] = (__bf16)v.x;
      As[mm * GEMM_BK + kk + 1] = (__bf16)v.y;
      As[mm * GEMM_BK + kk + 2] = (__bf16)v.z;
      As[mm * GEMM_BK + kk + 3] = (__bf16)v.w;
    }
#if defined(__AMDGCN__) && __has_builtin(__builtin_amdgcn_tensor_load_to_lds)
    if (w == 0) __builtin_amdgcn_s_wait_tensorcnt(0);
#endif
    __syncthreads();

    v16bf af[2];
    for (int mt = 0; mt < 2; ++mt)
      af[mt] = ld_frag(&As[(wm * 32 + mt * 16 + rw) * GEMM_BK], hi);
    for (int nt = 0; nt < 4; ++nt) {
      v16bf bf = ld_frag(&Bs[(wn * 64 + nt * 16 + rw) * GEMM_BK], hi);
      for (int mt = 0; mt < 2; ++mt)
        acc[mt][nt] = wmma_bf16(af[mt], bf, acc[mt][nt]);
    }
    __syncthreads();
  }

  // epilogue: C layout lanes0-15 M=r, lanes16-31 M=r+8; N = lane&15
  for (int mt = 0; mt < 2; ++mt)
    for (int nt = 0; nt < 4; ++nt)
      for (int r = 0; r < 8; ++r) {
        int gm = m0 + wm * 32 + mt * 16 + r + hi * 8;
        int gn = n0 + wn * 64 + nt * 16 + rw;
        if (gm >= M || gn >= Nn) continue;
        float v = acc[mt][nt][r];
        if (bias) v += bias[gn];
        float* cp = C + (size_t)gm * Nn + gn;
        if (mode == EPI_STORE)          *cp = v;
        else if (mode == EPI_ADD)       *cp += v;
        else if (mode == EPI_GELU)      *cp = gelu_tanh(v);
        else /* EPI_GATED_ADD */        *cp += gate[(size_t)gm * gstride] * v;
      }
}

// ---------------- embedding ----------------
__global__ __launch_bounds__(256) void embed_kernel(
    const int* __restrict__ ids, const float* __restrict__ tok,
    const float* __restrict__ pos, float* __restrict__ h)
{
  int row = blockIdx.x, tid = threadIdx.x;
  int t = ids[row];
  int s = row % kS;
  for (int j = 0; j < 4; ++j) {
    int d = tid + j * 256;
    h[(size_t)row * kD + d] = tok[(size_t)t * kD + d] + pos[(size_t)s * kD + d];
  }
}

// ---------------- layernorm (row = token, deterministic tree reductions) ------
__global__ __launch_bounds__(256) void layernorm_kernel(
    const float* __restrict__ x, const float* __restrict__ g,
    const float* __restrict__ bb, float* __restrict__ y)
{
  __shared__ float red[256];
  __shared__ float s_mean, s_rstd;
  int row = blockIdx.x, tid = threadIdx.x;
  const float* xr = x + (size_t)row * kD;
  float s = 0.f;
  for (int j = 0; j < 4; ++j) s += xr[tid + j * 256];
  red[tid] = s; __syncthreads();
  for (int st = 128; st > 0; st >>= 1) { if (tid < st) red[tid] += red[tid + st]; __syncthreads(); }
  if (tid == 0) s_mean = red[0] * (1.f / kD);
  __syncthreads();
  float m = s_mean;
  float v = 0.f;
  for (int j = 0; j < 4; ++j) { float d = xr[tid + j * 256] - m; v += d * d; }
  red[tid] = v; __syncthreads();
  for (int st = 128; st > 0; st >>= 1) { if (tid < st) red[tid] += red[tid + st]; __syncthreads(); }
  if (tid == 0) s_rstd = rsqrtf(red[0] * (1.f / kD) + 1e-5f);
  __syncthreads();
  float rs = s_rstd;
  for (int j = 0; j < 4; ++j) {
    int d = tid + j * 256;
    y[(size_t)row * kD + d] = (xr[d] - m) * rs * g[d] + bb[d];
  }
}

// ---------------- lightning indexer scores ----------------
// out[b,q,k] = sum_h hw[h] * relu( qi[b,q,h,:] . ki[b,k,h,:] )
__global__ __launch_bounds__(128) void idx_score_kernel(
    const float* __restrict__ qi, const float* __restrict__ ki,
    const float* __restrict__ hw, float* __restrict__ out)
{
  __shared__ __align__(16) __bf16 qs[32 * 64];
  __shared__ __align__(16) __bf16 kss[32 * 64];
  const int tid = threadIdx.x;
  const int q0 = blockIdx.x * 32, k0 = blockIdx.y * 32, b = blockIdx.z;
  const int w = tid >> 5, lane = tid & 31, hi = lane >> 4, rw = lane & 15;
  const int mt = w >> 1, nt = w & 1;

  float accv[8] = {0.f,0.f,0.f,0.f,0.f,0.f,0.f,0.f};
  for (int h = 0; h < kIH; ++h) {
    float hwv = hw[h];
    for (int r = 0; r < 16; ++r) {
      int lin = tid + r * 128;
      int row = lin >> 6, col = lin & 63;
      qs[row * 64 + col]  = (__bf16)qi[((size_t)(b * kS + q0 + row)) * (kIH * kID) + h * kID + col];
      kss[row * 64 + col] = (__bf16)ki[((size_t)(b * kS + k0 + row)) * (kIH * kID) + h * kID + col];
    }
    __syncthreads();
    v16bf a0 = ld_frag(qs  + (mt * 16 + rw) * 64,      hi);
    v16bf a1 = ld_frag(qs  + (mt * 16 + rw) * 64 + 32, hi);
    v16bf b0 = ld_frag(kss + (nt * 16 + rw) * 64,      hi);
    v16bf b1 = ld_frag(kss + (nt * 16 + rw) * 64 + 32, hi);
    v8f d = (v8f){0.f,0.f,0.f,0.f,0.f,0.f,0.f,0.f};
    d = wmma_bf16(a0, b0, d);
    d = wmma_bf16(a1, b1, d);
    for (int r = 0; r < 8; ++r) accv[r] += hwv * fmaxf(d[r], 0.f);
    __syncthreads();
  }
  for (int r = 0; r < 8; ++r) {
    int m = mt * 16 + r + hi * 8, n = nt * 16 + rw;
    out[((size_t)(b * kS + q0 + m)) * kS + k0 + n] = accv[r];
  }
}

// ---------------- exact top-k(512) per row -> selection bitmask ----------------
__global__ __launch_bounds__(256) void topk_kernel(
    const float* __restrict__ scores, unsigned* __restrict__ sel, int ksel)
{
  __shared__ int s_cnt;
  __shared__ int ec[256];
  __shared__ unsigned selw[32];
  int row = blockIdx.x, tid = threadIdx.x;
  const float* sr = scores + (size_t)row * kS;

  unsigned key[4];
  for (int j = 0; j < 4; ++j) {
    unsigned u = __float_as_uint(sr[tid * 4 + j]);
    key[j] = (u & 0x80000000u) ? ~u : (u | 0x80000000u);   // order-preserving
  }
  unsigned T = 0;
  for (int bit = 31; bit >= 0; --bit) {
    unsigned cand = T | (1u << bit);
    if (tid == 0) s_cnt = 0;
    __syncthreads();
    int c = 0;
    for (int j = 0; j < 4; ++j) c += (key[j] >= cand) ? 1 : 0;
    atomicAdd(&s_cnt, c);
    __syncthreads();
    if (s_cnt >= ksel) T = cand;
    __syncthreads();
  }
  if (tid == 0) s_cnt = 0;
  __syncthreads();
  { int c = 0; for (int j = 0; j < 4; ++j) c += (key[j] > T) ? 1 : 0; atomicAdd(&s_cnt, c); }
  __syncthreads();
  int need_eq = ksel - s_cnt;
  int ce = 0;
  for (int j = 0; j < 4; ++j) ce += (key[j] == T) ? 1 : 0;
  ec[tid] = ce; __syncthreads();
  if (tid == 0) { int acc = 0; for (int i = 0; i < 256; ++i) { int t = ec[i]; ec[i] = acc; acc += t; } }
  __syncthreads();
  if (tid < 32) selw[tid] = 0u;
  __syncthreads();
  int base = ec[tid];
  for (int j = 0; j < 4; ++j) {
    int idx = tid * 4 + j;
    bool pick = false;
    if (key[j] > T) pick = true;
    else if (key[j] == T) { pick = (base < need_eq); base++; }
    if (pick) atomicOr(&selw[idx >> 5], 1u << (idx & 31));
  }
  __syncthreads();
  if (tid < 32) sel[(size_t)row * 32 + tid] = selw[tid];
}

// ---------------- fused attention (scores in LDS, two-pass softmax) ------------
// grid: (B*H, S/32); block 256; dynamic LDS:
//   qs 32x64 bf16 | ks/vs 64x64 bf16 | ps 32x1024 bf16 | sc 32x1024 f32
__global__ __launch_bounds__(256) void attention_kernel(
    const float* __restrict__ Q, const float* __restrict__ Kb,
    const float* __restrict__ Vb, const unsigned* __restrict__ sel,
    float* __restrict__ O)
{
  extern __shared__ char smem[];
  __bf16* qs = (__bf16*)smem;                          // 4096 B
  __bf16* ks = (__bf16*)(smem + 4096);                 // 8192 B (reused for V^T)
  __bf16* ps = (__bf16*)(smem + 4096 + 8192);          // 65536 B
  float*  sc = (float*)(smem + 4096 + 8192 + 65536);   // 131072 B

  const int tid = threadIdx.x;
  const int bh = blockIdx.x;
  const int b = bh / kH, hh = bh % kH;
  const int q0 = blockIdx.y * 32;
  const int w = tid >> 5, lane = tid & 31, hi = lane >> 4, rw = lane & 15;
  const int mt = w >> 2, nt = w & 3;    // 2 x 4 wave grid over 32x64 output

  for (int r = 0; r < 8; ++r) {
    int lin = tid + r * 256;
    int row = lin >> 6, col = lin & 63;
    qs[row * 64 + col] =
        (__bf16)(Q[((size_t)(b * kS + q0 + row)) * kD + hh * kDH + col] * 0.125f);
  }
  __syncthreads();

  for (int kt = 0; kt < 16; ++kt) {
    int k0 = kt * 64;
    for (int r = 0; r < 16; ++r) {
      int lin = tid + r * 256;
      int row = lin >> 6, col = lin & 63;
      ks[row * 64 + col] =
          (__bf16)Kb[((size_t)(b * kS + k0 + row)) * kD + hh * kDH + col];
    }
    __syncthreads();
    v16bf a0 = ld_frag(qs + (mt * 16 + rw) * 64,      hi);
    v16bf a1 = ld_frag(qs + (mt * 16 + rw) * 64 + 32, hi);
    v16bf b0 = ld_frag(ks + (nt * 16 + rw) * 64,      hi);
    v16bf b1 = ld_frag(ks + (nt * 16 + rw) * 64 + 32, hi);
    v8f d = (v8f){0.f,0.f,0.f,0.f,0.f,0.f,0.f,0.f};
    d = wmma_bf16(a0, b0, d);
    d = wmma_bf16(a1, b1, d);
    for (int r = 0; r < 8; ++r) {
      int m = mt * 16 + r + hi * 8;
      int n = nt * 16 + rw;
      int qg = q0 + m, kg = k0 + n;
      bool ok = (kg <= qg) &&
                (((sel[((size_t)(b * kS + qg)) * 32 + (kg >> 5)] >> (kg & 31)) & 1u) != 0u);
      sc[m * kS + kg] = d[r] + (ok ? 0.f : kNEG);
    }
    __syncthreads();
  }

  {
    int row = tid >> 3, sub = tid & 7;
    float mx = -3.4e38f;
    for (int c = sub; c < kS; c += 8) mx = fmaxf(mx, sc[row * kS + c]);
    for (int off = 4; off >= 1; off >>= 1) mx = fmaxf(mx, __shfl_xor(mx, off, 8));
    float sum = 0.f;
    for (int c = sub; c < kS; c += 8) sum += __expf(sc[row * kS + c] - mx);
    for (int off = 4; off >= 1; off >>= 1) sum += __shfl_xor(sum, off, 8);
    float inv = 1.0f / sum;
    for (int c = sub; c < kS; c += 8)
      ps[row * kS + c] = (__bf16)(__expf(sc[row * kS + c] - mx) * inv);
  }
  __syncthreads();

  v8f o = (v8f){0.f,0.f,0.f,0.f,0.f,0.f,0.f,0.f};
  for (int kt = 0; kt < 16; ++kt) {
    for (int r = 0; r < 16; ++r) {
      int lin = tid + r * 256;
      int tok = lin >> 6, dh = lin & 63;
      ks[dh * 64 + tok] =
          (__bf16)Vb[((size_t)(b * kS + kt * 64 + tok)) * kD + hh * kDH + dh];
    }
    __syncthreads();
    v16bf a0 = ld_frag(ps + (mt * 16 + rw) * kS + kt * 64,      hi);
    v16bf a1 = ld_frag(ps + (mt * 16 + rw) * kS + kt * 64 + 32, hi);
    v16bf b0 = ld_frag(ks + (nt * 16 + rw) * 64,      hi);
    v16bf b1 = ld_frag(ks + (nt * 16 + rw) * 64 + 32, hi);
    o = wmma_bf16(a0, b0, o);
    o = wmma_bf16(a1, b1, o);
    __syncthreads();
  }
  for (int r = 0; r < 8; ++r) {
    int m = mt * 16 + r + hi * 8;
    int n = nt * 16 + rw;
    O[((size_t)(b * kS + q0 + m)) * kD + hh * kDH + n] = o[r];
  }
}

// ---------------- router: softmax + top-2 gates ----------------
__global__ __launch_bounds__(256) void router_kernel(
    const float* __restrict__ x, const float* __restrict__ rwt,
    const float* __restrict__ rb, float* __restrict__ gates)
{
  __shared__ float red[256 * kE];
  int n = blockIdx.x, tid = threadIdx.x;
  float p[kE] = {0.f,0.f,0.f,0.f,0.f,0.f,0.f,0.f};
  const float* xr = x + (size_t)n * kD;
  for (int d = tid; d < kD; d += 256) {
    float xv = xr[d];
    const float* wr = rwt + (size_t)d * kE;
    for (int e = 0; e < kE; ++e) p[e] += xv * wr[e];
  }
  for (int e = 0; e < kE; ++e) red[tid * kE + e] = p[e];
  __syncthreads();
  for (int st = 128; st > 0; st >>= 1) {
    if (tid < st)
      for (int e = 0; e < kE; ++e) red[tid * kE + e] += red[(tid + st) * kE + e];
    __syncthreads();
  }
  if (tid == 0) {
    float lg[kE], mx = -3.4e38f;
    for (int e = 0; e < kE; ++e) { lg[e] = red[e] + rb[e]; mx = fmaxf(mx, lg[e]); }
    float sum = 0.f;
    for (int e = 0; e < kE; ++e) { lg[e] = __expf(lg[e] - mx); sum += lg[e]; }
    for (int e = 0; e < kE; ++e) lg[e] /= sum;
    int i1 = 0;
    for (int e = 1; e < kE; ++e) if (lg[e] > lg[i1]) i1 = e;
    int i2 = -1;
    for (int e = 0; e < kE; ++e) { if (e == i1) continue; if (i2 < 0 || lg[e] > lg[i2]) i2 = e; }
    float tv = lg[i1] + lg[i2];
    for (int e = 0; e < kE; ++e) gates[(size_t)n * kE + e] = 0.f;
    gates[(size_t)n * kE + i1] = lg[i1] / tv;
    gates[(size_t)n * kE + i2] = lg[i2] / tv;
  }
}

// ---------------- host orchestration ----------------
static inline void gemm_conv(const float* A, const float* W, const float* bias,
                             float* C, const float* gate, int gstride,
                             int M, int Nn, int K, int mode,
                             __bf16* wt, hipStream_t stream) {
  dim3 tgrid((Nn + 31) / 32, (K + 31) / 32);
  transpose_to_bf16_kernel<<<tgrid, 256, 0, stream>>>(W, wt, K, Nn);
  dim3 grid((Nn + GEMM_BN - 1) / GEMM_BN, (M + GEMM_BM - 1) / GEMM_BM);
  gemm_bf16_kernel<<<grid, 256, 0, stream>>>(A, wt, bias, C, gate, gstride, M, Nn, K, mode);
}

extern "C" void kernel_launch(void* const* d_in, const int* in_sizes, int n_in,
                              void* d_out, int out_size, void* d_ws, size_t ws_size,
                              hipStream_t stream) {
  (void)in_sizes; (void)n_in; (void)out_size; (void)ws_size;

  const int*   input_ids = (const int*)d_in[0];
  const float* tok_emb   = (const float*)d_in[1];
  const float* pos_emb   = (const float*)d_in[2];
  const float* ind_qw    = (const float*)d_in[3];
  const float* ind_qb    = (const float*)d_in[4];
  const float* ind_kw    = (const float*)d_in[5];
  const float* ind_kb    = (const float*)d_in[6];
  const float* ind_hw    = (const float*)d_in[7];
  const float* an_g      = (const float*)d_in[8];
  const float* an_b      = (const float*)d_in[9];
  const float* q_w       = (const float*)d_in[10];
  const float* q_b       = (const float*)d_in[11];
  const float* k_w       = (const float*)d_in[12];
  const float* k_b       = (const float*)d_in[13];
  const float* v_w       = (const float*)d_in[14];
  const float* v_b       = (const float*)d_in[15];
  const float* o_w       = (const float*)d_in[16];
  const float* o_b       = (const float*)d_in[17];
  const float* mn_g      = (const float*)d_in[18];
  const float* mn_b      = (const float*)d_in[19];
  const float* router_w  = (const float*)d_in[20];
  const float* router_b  = (const float*)d_in[21];
  const float* e_w1      = (const float*)d_in[22];
  const float* e_b1      = (const float*)d_in[23];
  const float* e_w2      = (const float*)d_in[24];
  const float* e_b2      = (const float*)d_in[25];
  const float* ln_g      = (const float*)d_in[26];
  const float* ln_b      = (const float*)d_in[27];
  const float* out_w     = (const float*)d_in[28];
  const float* out_b     = (const float*)d_in[29];
  float* logits = (float*)d_out;

  char* wsb = (char*)d_ws;
  size_t off = 0;
  auto take = [&](size_t bytes) -> char* {
    char* p = wsb + off;
    off = (off + bytes + 255) & ~(size_t)255;
    return p;
  };
  float*    h    = (float*)take((size_t)kN * kD * 4);
  float*    hn   = (float*)take((size_t)kN * kD * 4);
  float*    qbuf = (float*)take((size_t)kN * kD * 4);
  float*    kbuf = (float*)take((size_t)kN * kD * 4);
  float*    vbuf = (float*)take((size_t)kN * kD * 4);
  float*    ao   = (float*)take((size_t)kN * kD * 4);
  float*    qi   = (float*)take((size_t)kN * kIH * kID * 4);
  float*    ki   = (float*)take((size_t)kN * kIH * kID * 4);
  float*    isc  = (float*)take((size_t)kB * kS * kS * 4);
  unsigned* sel  = (unsigned*)take((size_t)kB * kS * (kS / 32) * 4);
  float*    gate = (float*)take((size_t)kN * kE * 4);
  float*    y1   = (float*)take((size_t)kN * kFF * 4);
  float*    mxn  = (float*)take((size_t)kN * kD * 4);
  __bf16*   wt   = (__bf16*)take((size_t)kV * kD * 2);   // max-size bf16 W^T scratch

  const size_t attn_lds = 4096 + 8192 + 65536 + 131072;   // ~204 KB of the 320 KB WGP LDS
  (void)hipFuncSetAttribute(reinterpret_cast<const void*>(attention_kernel),
                            hipFuncAttributeMaxDynamicSharedMemorySize, (int)attn_lds);

  embed_kernel<<<kN, 256, 0, stream>>>(input_ids, tok_emb, pos_emb, h);

  for (int l = 0; l < kL; ++l) {
    // lightning indexer projections (from raw h) + scores + top-k mask
    gemm_conv(h, ind_qw + (size_t)l * kD * kIH * kID, ind_qb + (size_t)l * kIH * kID,
              qi, nullptr, 0, kN, kIH * kID, kD, EPI_STORE, wt, stream);
    gemm_conv(h, ind_kw + (size_t)l * kD * kIH * kID, ind_kb + (size_t)l * kIH * kID,
              ki, nullptr, 0, kN, kIH * kID, kD, EPI_STORE, wt, stream);
    idx_score_kernel<<<dim3(kS / 32, kS / 32, kB), 128, 0, stream>>>(
        qi, ki, ind_hw + (size_t)l * kIH, isc);
    topk_kernel<<<kB * kS, 256, 0, stream>>>(isc, sel, kS / 2);

    // attention
    layernorm_kernel<<<kN, 256, 0, stream>>>(h, an_g + (size_t)l * kD, an_b + (size_t)l * kD, hn);
    gemm_conv(hn, q_w + (size_t)l * kD * kD, q_b + (size_t)l * kD, qbuf, nullptr, 0,
              kN, kD, kD, EPI_STORE, wt, stream);
    gemm_conv(hn, k_w + (size_t)l * kD * kD, k_b + (size_t)l * kD, kbuf, nullptr, 0,
              kN, kD, kD, EPI_STORE, wt, stream);
    gemm_conv(hn, v_w + (size_t)l * kD * kD, v_b + (size_t)l * kD, vbuf, nullptr, 0,
              kN, kD, kD, EPI_STORE, wt, stream);
    attention_kernel<<<dim3(kB * kH, kS / 32), 256, attn_lds, stream>>>(
        qbuf, kbuf, vbuf, sel, ao);
    gemm_conv(ao, o_w + (size_t)l * kD * kD, o_b + (size_t)l * kD, h, nullptr, 0,
              kN, kD, kD, EPI_ADD, wt, stream);   // h += ao @ o_w + o_b

    // MoE: dense per-expert, gate-weighted accumulate straight into h
    layernorm_kernel<<<kN, 256, 0, stream>>>(h, mn_g + (size_t)l * kD, mn_b + (size_t)l * kD, mxn);
    router_kernel<<<kN, 256, 0, stream>>>(mxn, router_w + (size_t)l * kD * kE,
                                          router_b + (size_t)l * kE, gate);
    for (int e = 0; e < kE; ++e) {
      const float* w1 = e_w1 + ((size_t)l * kE + e) * kD * kFF;
      const float* b1 = e_b1 + ((size_t)l * kE + e) * kFF;
      const float* w2 = e_w2 + ((size_t)l * kE + e) * kFF * kD;
      const float* b2 = e_b2 + ((size_t)l * kE + e) * kD;
      gemm_conv(mxn, w1, b1, y1, nullptr, 0, kN, kFF, kD, EPI_GELU, wt, stream);
      gemm_conv(y1, w2, b2, h, gate + e, kE, kN, kD, kFF, EPI_GATED_ADD, wt, stream);
    }
  }

  // final LN + logits
  layernorm_kernel<<<kN, 256, 0, stream>>>(h, ln_g, ln_b, hn);
  gemm_conv(hn, out_w, out_b, logits, nullptr, 0, kN, kV, kD, EPI_STORE, wt, stream);
}
